// MHA_19971597926464
// MI455X (gfx1250) — matrix operand order repaired and checked by reading, not compiled, not run on hardware
//
#include <hip/hip_runtime.h>
#include <hip/hip_bf16.h>
#include <stdint.h>

// ---------------------------------------------------------------------------
// MHA for MI455X (gfx1250): all matmuls on v_wmma_f32_16x16x32_f16 (wave32).
//   1) Q/K/V projections: f32 GEMM -> f16 workspace
//   2) flash attention (streaming softmax, WMMA for QK^T and P@V,
//      Q/K tiles staged with GLOBAL_LOAD_ASYNC_TO_LDS_B128)
//   3) output projection: f16 ctx @ Wo^T -> f32 d_out
// ---------------------------------------------------------------------------

typedef _Float16 half4v  __attribute__((ext_vector_type(4)));
typedef _Float16 half8v  __attribute__((ext_vector_type(8)));
typedef _Float16 half16v __attribute__((ext_vector_type(16)));
typedef float    float4v __attribute__((ext_vector_type(4)));
typedef float    float8v __attribute__((ext_vector_type(8)));

#define B_  4
#define LQ_ 2048
#define LV_ 2048
#define D_  1024
#define H_  16
#define DH_ 64

// ---------------------------------------------------------------------------
// WMMA fragment helpers (layouts per CDNA5 ISA 7.12.2, wave32):
//  A 16x32 f16:  lane L: m = L%16, half = L/16
//     elements 0..7  hold K = kBase +      half*8 + e
//     elements 8..15 hold K = kBase + 16 + half*8 + (e-8)
//  B 32x16 f16:  lane L: n = L%16, half = L/16
//     elements 0..15 hold K in the same order as A
//  C/D 16x16 f32: lane L: n = L%16, element e: m = (L/16)*8 + e
// LDS tiles are row-major [row][k] with padded stride (16B-aligned rows).
// ---------------------------------------------------------------------------

__device__ __forceinline__ half16v pack16(half8v lo, half8v hi) {
    half16v r;
#pragma unroll
    for (int i = 0; i < 8; ++i) { r[i] = lo[i]; r[i + 8] = hi[i]; }
    return r;
}

__device__ __forceinline__ half16v frag_a_lds(const _Float16* tile, int ldk,
                                              int mBase, int kBase, int lane) {
    const int m = mBase + (lane & 15);
    const int hh = lane >> 4;
    const _Float16* p = tile + m * ldk + kBase + hh * 8;
    half8v lo = *(const half8v*)(p);        // K = kBase + hh*8 + 0..7
    half8v hi = *(const half8v*)(p + 16);   // K = kBase + 16 + hh*8 + 0..7
    return pack16(lo, hi);
}

__device__ __forceinline__ half16v frag_b_lds(const _Float16* tile, int ldk,
                                              int nBase, int kBase, int lane) {
    const int n = nBase + (lane & 15);
    const int hh = lane >> 4;
    const _Float16* p = tile + n * ldk + kBase + hh * 8;
    half8v lo = *(const half8v*)(p);        // K = kBase + hh*8 + 0..7
    half8v hi = *(const half8v*)(p + 16);   // K = kBase + 16 + hh*8 + 0..7
    return pack16(lo, hi);
}

__device__ __forceinline__ float8v wmma_f16(half16v a, half16v b, float8v c) {
    return __builtin_amdgcn_wmma_f32_16x16x32_f16(
        /*neg_a=*/false, a, /*neg_b=*/false, b,
        /*c_mod=*/(short)0, c, /*reuse_a=*/false, /*reuse_b=*/false);
}

// --- gfx1250 async global->LDS copy (ASYNCcnt-tracked, 16B per lane) -------
__device__ __forceinline__ void async_copy_b128(uint32_t lds_byte_addr,
                                                const void* gaddr) {
    asm volatile("global_load_async_to_lds_b128 %0, %1, off"
                 :: "v"(lds_byte_addr),
                    "v"((unsigned long long)(uintptr_t)gaddr)
                 : "memory");
}
__device__ __forceinline__ void wait_async0() {
    asm volatile("s_wait_asynccnt 0x0" ::: "memory");
}
__device__ __forceinline__ uint32_t lds_addr_u32(const void* p) {
    // flat->LDS mapping truncates to addr[31:0] (CDNA5 ISA 10.2)
    return (uint32_t)(uintptr_t)p;
}

// ---------------------------------------------------------------------------
// GEMM: C[M,N] = A[M,K] @ W[N,K]^T   (nn.Linear semantics)
// Tiles: BM=128, BN=64, BK=32. 256 threads = 8 waves, waves 4(M) x 2(N),
// each wave owns a 32x32 register accumulator -> 4 WMMAs per K-step.
// ---------------------------------------------------------------------------
template <bool A_HALF, bool OUT_F32>
__global__ __launch_bounds__(256) void linear_wmma_kernel(
    const void* __restrict__ Aw, const float* __restrict__ W,
    void* __restrict__ Cw, int M, int N, int K)
{
    constexpr int BM = 128, BN = 64, BK = 32, LDT = 40;  // LDT pad -> 80B rows
    __shared__ _Float16 As[BM * LDT];
    __shared__ _Float16 Ws[BN * LDT];

    const int tid  = threadIdx.x;
    const int lane = tid & 31;
    const int w    = tid >> 5;
    const int wm   = w & 3;   // 4 waves along M
    const int wn   = w >> 2;  // 2 waves along N
    const int mBlock = blockIdx.y * BM;
    const int nBlock = blockIdx.x * BN;

    const float*    Af = (const float*)Aw;
    const _Float16* Ah = (const _Float16*)Aw;

    const float8v zero = {};
    float8v acc[2][2];
    acc[0][0] = zero; acc[0][1] = zero; acc[1][0] = zero; acc[1][1] = zero;

    const int c4 = (tid & 7) * 4;   // 8 threads cover 32 cols, 4 elems each
    const int ra = tid >> 3;        // 0..31

    for (int k0 = 0; k0 < K; k0 += BK) {
        __syncthreads();
        // --- stage A tile (BM x BK) as f16 ---
#pragma unroll
        for (int j = 0; j < 4; ++j) {
            const int r = ra + j * 32;
            const size_t gi = (size_t)(mBlock + r) * K + k0 + c4;
            half4v hv;
            if constexpr (A_HALF) {
                hv = *(const half4v*)(Ah + gi);
                if (k0 + BK < K) __builtin_prefetch((const void*)(Ah + gi + BK), 0, 1);
            } else {
                float4v fv = *(const float4v*)(Af + gi);
#pragma unroll
                for (int i = 0; i < 4; ++i) hv[i] = (_Float16)fv[i];
                if (k0 + BK < K) __builtin_prefetch((const void*)(Af + gi + BK), 0, 1);
            }
            *(half4v*)(&As[r * LDT + c4]) = hv;
        }
        // --- stage W tile (BN x BK) as f16 ---
#pragma unroll
        for (int j = 0; j < 2; ++j) {
            const int r = ra + j * 32;
            const size_t gi = (size_t)(nBlock + r) * K + k0 + c4;
            float4v fv = *(const float4v*)(W + gi);
            half4v hv;
#pragma unroll
            for (int i = 0; i < 4; ++i) hv[i] = (_Float16)fv[i];
            *(half4v*)(&Ws[r * LDT + c4]) = hv;
        }
        __syncthreads();

        // --- compute: 32x32 per wave ---
        half16v a0 = frag_a_lds(As, LDT, wm * 32 + 0,  0, lane);
        half16v a1 = frag_a_lds(As, LDT, wm * 32 + 16, 0, lane);
        half16v b0 = frag_b_lds(Ws, LDT, wn * 32 + 0,  0, lane);
        half16v b1 = frag_b_lds(Ws, LDT, wn * 32 + 16, 0, lane);
        acc[0][0] = wmma_f16(a0, b0, acc[0][0]);
        acc[0][1] = wmma_f16(a0, b1, acc[0][1]);
        acc[1][0] = wmma_f16(a1, b0, acc[1][0]);
        acc[1][1] = wmma_f16(a1, b1, acc[1][1]);
    }

    // --- epilogue ---
    const int cn  = lane & 15;
    const int cm0 = (lane >> 4) * 8;
#pragma unroll
    for (int mi = 0; mi < 2; ++mi)
#pragma unroll
    for (int ni = 0; ni < 2; ++ni) {
        const int gm0 = mBlock + wm * 32 + mi * 16 + cm0;
        const int gn  = nBlock + wn * 32 + ni * 16 + cn;
#pragma unroll
        for (int e = 0; e < 8; ++e) {
            const size_t gi = (size_t)(gm0 + e) * N + gn;
            if constexpr (OUT_F32) ((float*)Cw)[gi] = acc[mi][ni][e];
            else                   ((_Float16*)Cw)[gi] = (_Float16)acc[mi][ni][e];
        }
    }
}

// ---------------------------------------------------------------------------
// Flash attention. Grid: (LQ/64, B*H). Block: 256 threads = 8 waves.
// Q and K tiles staged via async global->LDS (no VGPR round-trip); V tile
// loaded b128 and transposed into LDS; scores scaled by 1/sqrt(DH) at
// write-out; O accumulators live in registers across the KV loop.
// ---------------------------------------------------------------------------
__global__ __launch_bounds__(256) void flash_attn_kernel(
    const _Float16* __restrict__ Qp, const _Float16* __restrict__ Kp,
    const _Float16* __restrict__ Vp, _Float16* __restrict__ Ctx)
{
    constexpr int BQ = 64, BKV = 64, LDT = 72, LDS32 = 66;
    __shared__ _Float16 Qs [BQ  * LDT];   // [q][d]
    __shared__ _Float16 Ks [BKV * LDT];   // [kv][d]
    __shared__ _Float16 Vts[DH_ * LDT];   // [d][kv]  (transposed V)
    __shared__ _Float16 Ps [BQ  * LDT];   // [q][kv]  softmax probs (f16)
    __shared__ float    Sf [BQ  * LDS32]; // [q][kv]  raw scores (f32)
    __shared__ float    mRow[BQ], lRow[BQ], scRow[BQ];

    const int tid  = threadIdx.x;
    const int lane = tid & 31;
    const int w    = tid >> 5;
    const int bh   = blockIdx.y;
    const int b    = bh >> 4;   // / H
    const int h    = bh & 15;   // % H
    const int qBase = blockIdx.x * BQ;

    const _Float16* Qg = Qp + ((size_t)b * LQ_ + qBase) * D_ + h * DH_;
    const _Float16* Kg = Kp + ((size_t)b * LV_) * D_ + h * DH_;
    const _Float16* Vg = Vp + ((size_t)b * LV_) * D_ + h * DH_;

    // 64x64 f16 tile = 512 16B-chunks; 256 threads -> 2 async copies each.
    // chunk ch: row r = ch>>3, f16 col c8 = (ch&7)*8.
    {
#pragma unroll
        for (int j = 0; j < 2; ++j) {
            const int ch = tid + j * 256;
            const int r  = ch >> 3;
            const int c8 = (ch & 7) * 8;
            async_copy_b128(lds_addr_u32(&Qs[r * LDT + c8]),
                            (const void*)(Qg + (size_t)r * D_ + c8));
        }
    }
    if (tid < BQ) { mRow[tid] = -1e30f; lRow[tid] = 0.0f; }

    const float8v zero = {};
    float8v accO[2]; accO[0] = zero; accO[1] = zero;

    for (int kv = 0; kv < LV_; kv += BKV) {
        __syncthreads();  // previous iteration's LDS reads complete
        // --- stage K tile via async copy ---
#pragma unroll
        for (int j = 0; j < 2; ++j) {
            const int ch = tid + j * 256;
            const int r  = ch >> 3;
            const int c8 = (ch & 7) * 8;
            async_copy_b128(lds_addr_u32(&Ks[r * LDT + c8]),
                            (const void*)(Kg + (size_t)(kv + r) * D_ + c8));
        }
        // --- stage transposed V tile (b128 loads, scalar transposed stores) ---
        {
            const int c8 = (tid & 7) * 8;
            const int r0 = tid >> 3;    // 0..31
#pragma unroll
            for (int j = 0; j < 2; ++j) {
                const int r = r0 + j * 32;
                half8v vv = *(const half8v*)(Vg + (size_t)(kv + r) * D_ + c8);
#pragma unroll
                for (int i = 0; i < 8; ++i) Vts[(c8 + i) * LDT + r] = vv[i];
            }
        }
        wait_async0();      // Q (first iter) + K tile resident in LDS
        __syncthreads();

        // --- S = Q @ K^T (each wave: 2 of 16 score tiles) ---
#pragma unroll
        for (int i = 0; i < 2; ++i) {
            const int st = w * 2 + i;
            const int qt = st >> 2, kt = st & 3;
            float8v accS = zero;
            accS = wmma_f16(frag_a_lds(Qs, LDT, qt * 16, 0,  lane),
                            frag_b_lds(Ks, LDT, kt * 16, 0,  lane), accS);
            accS = wmma_f16(frag_a_lds(Qs, LDT, qt * 16, 32, lane),
                            frag_b_lds(Ks, LDT, kt * 16, 32, lane), accS);
            const int n  = kt * 16 + (lane & 15);
            const int m0 = qt * 16 + ((lane >> 4) << 3);
#pragma unroll
            for (int e = 0; e < 8; ++e)
                Sf[(m0 + e) * LDS32 + n] = accS[e] * 0.125f;  // 1/sqrt(64)
        }
        __syncthreads();

        // --- online softmax, one thread per q row ---
        if (tid < BQ) {
            const int q = tid;
            const float mOld = mRow[q];
            float mNew = mOld;
#pragma unroll 8
            for (int c = 0; c < BKV; ++c) mNew = fmaxf(mNew, Sf[q * LDS32 + c]);
            float sum = 0.0f;
#pragma unroll 8
            for (int c = 0; c < BKV; ++c) {
                const float p = __expf(Sf[q * LDS32 + c] - mNew);
                sum += p;
                Ps[q * LDT + c] = (_Float16)p;
            }
            const float sc = __expf(mOld - mNew);
            lRow[q]  = lRow[q] * sc + sum;
            mRow[q]  = mNew;
            scRow[q] = sc;
        }
        __syncthreads();

        // --- O = diag(sc)*O + P @ V (each wave: 2 of 16 output tiles) ---
#pragma unroll
        for (int i = 0; i < 2; ++i) {
            const int ot = w * 2 + i;
            const int qt = ot >> 2, dt = ot & 3;
            const int m0 = qt * 16 + ((lane >> 4) << 3);
#pragma unroll
            for (int e = 0; e < 8; ++e) accO[i][e] *= scRow[m0 + e];
            accO[i] = wmma_f16(frag_a_lds(Ps,  LDT, qt * 16, 0,  lane),
                               frag_b_lds(Vts, LDT, dt * 16, 0,  lane), accO[i]);
            accO[i] = wmma_f16(frag_a_lds(Ps,  LDT, qt * 16, 32, lane),
                               frag_b_lds(Vts, LDT, dt * 16, 32, lane), accO[i]);
        }
    }

    // --- finalize: O /= l, store context as f16 in [b][l][h*DH+d] layout ---
#pragma unroll
    for (int i = 0; i < 2; ++i) {
        const int ot = w * 2 + i;
        const int qt = ot >> 2, dt = ot & 3;
        const int m0 = qt * 16 + ((lane >> 4) << 3);
        const int n  = dt * 16 + (lane & 15);
#pragma unroll
        for (int e = 0; e < 8; ++e) {
            const float o = accO[i][e] / lRow[m0 + e];
            Ctx[((size_t)b * LQ_ + qBase + m0 + e) * D_ + h * DH_ + n] = (_Float16)o;
        }
    }
}

// ---------------------------------------------------------------------------
extern "C" void kernel_launch(void* const* d_in, const int* in_sizes, int n_in,
                              void* d_out, int out_size, void* d_ws, size_t ws_size,
                              hipStream_t stream) {
    (void)in_sizes; (void)n_in; (void)out_size; (void)ws_size;
    const float* q  = (const float*)d_in[0];
    const float* k  = (const float*)d_in[1];
    const float* v  = (const float*)d_in[2];
    const float* Wq = (const float*)d_in[3];
    const float* Wk = (const float*)d_in[4];
    const float* Wv = (const float*)d_in[5];
    const float* Wo = (const float*)d_in[6];
    float* out = (float*)d_out;

    // workspace: Qp | Kp | Vp | Ctx, all f16 [B, L, D]  (64 MB total)
    const size_t NE = (size_t)B_ * LQ_ * D_;
    _Float16* Qp  = (_Float16*)d_ws;
    _Float16* Kp  = Qp + NE;
    _Float16* Vp  = Kp + NE;
    _Float16* Ctx = Vp + NE;

    dim3 blk(256);
    dim3 gemmGrid(D_ / 64, (B_ * LQ_) / 128);  // (N/BN, M/BM)

    linear_wmma_kernel<false, false><<<gemmGrid, blk, 0, stream>>>(q, Wq, Qp, B_ * LQ_, D_, D_);
    linear_wmma_kernel<false, false><<<gemmGrid, blk, 0, stream>>>(k, Wk, Kp, B_ * LV_, D_, D_);
    linear_wmma_kernel<false, false><<<gemmGrid, blk, 0, stream>>>(v, Wv, Vp, B_ * LV_, D_, D_);

    dim3 faGrid(LQ_ / 64, B_ * H_);
    flash_attn_kernel<<<faGrid, blk, 0, stream>>>(Qp, Kp, Vp, Ctx);

    linear_wmma_kernel<true, true><<<gemmGrid, blk, 0, stream>>>(Ctx, Wo, out, B_ * LQ_, D_, D_);
}
